// MLP_Binary_9869834846708
// MI455X (gfx1250) — matrix operand order repaired and checked by reading, not compiled, not run on hardware
//
#include <hip/hip_runtime.h>
#include <hip/hip_bf16.h>

typedef __attribute__((ext_vector_type(16))) _Float16 v16h;
typedef __attribute__((ext_vector_type(8)))  _Float16 v8h;
typedef __attribute__((ext_vector_type(8)))  float    v8f;

#define B_ROWS   65536
#define D_IN     784
#define K_PAD    800      // 784 padded to multiple of 32
#define D_H      1024
#define N_CLS    10

// ---------------- prep kernels ----------------

__global__ void prep_x_kernel(const float* __restrict__ x, _Float16* __restrict__ xb) {
    int b = blockIdx.x;                       // 0..65535
    int k = blockIdx.y * blockDim.x + threadIdx.x;
    if (k < K_PAD) {
        _Float16 v = (_Float16)0.f;
        if (k < D_IN) v = (_Float16)x[(size_t)b * D_IN + k];
        xb[(size_t)b * K_PAD + k] = v;
    }
}

__global__ void prep_w1_kernel(const float* __restrict__ w1, _Float16* __restrict__ w1b) {
    int j = blockIdx.x;                       // 0..1023
    int k = blockIdx.y * blockDim.x + threadIdx.x;
    if (k < K_PAD) {
        _Float16 v = (_Float16)0.f;
        if (k < D_IN) v = (w1[(size_t)j * D_IN + k] >= 0.f) ? (_Float16)1.f : (_Float16)(-1.f);
        w1b[(size_t)j * K_PAD + k] = v;
    }
}

__global__ void prep_w2_kernel(const float* __restrict__ w2, _Float16* __restrict__ w2b) {
    int c = blockIdx.x;                       // 0..15 (padded classes)
    int j = blockIdx.y * blockDim.x + threadIdx.x;
    if (j < D_H) {
        _Float16 v = (_Float16)0.f;
        if (c < N_CLS) v = (w2[(size_t)c * D_H + j] >= 0.f) ? (_Float16)1.f : (_Float16)(-1.f);
        w2b[(size_t)c * D_H + j] = v;
    }
}

__global__ void zero_stats_kernel(float* __restrict__ colsum, float* __restrict__ colsq) {
    int i = blockIdx.x * blockDim.x + threadIdx.x;
    if (i < D_H) { colsum[i] = 0.f; colsq[i] = 0.f; }
}

// ---------------- GEMM1: h = x @ sign(w1)^T, + per-column stats ----------------
// Wave tile: 32M x 64N (2 M-subtiles x 4 N-subtiles). Block = 8 waves along N
// -> block covers 32M x 512N. Grid (2048, 2).

union AFrag { v16h v; v8h h2[2]; };

__global__ __launch_bounds__(256)
void gemm1_wmma_kernel(const _Float16* __restrict__ xb, const _Float16* __restrict__ w1b,
                       float* __restrict__ h, float* __restrict__ colsum,
                       float* __restrict__ colsq)
{
    const int lane = threadIdx.x & 31;
    const int wave = threadIdx.x >> 5;
    const int l15  = lane & 15;
    const int hi   = lane >> 4;
    const int m0   = blockIdx.x * 32;
    const int n0   = blockIdx.y * 512 + wave * 64;

    v8f acc[2][4];
    #pragma unroll
    for (int mt = 0; mt < 2; ++mt)
        #pragma unroll
        for (int nt = 0; nt < 4; ++nt)
            acc[mt][nt] = {};

    // A fragment: row = m0 + mt*16 + l15; per K-step loads K = kk+hi*8..+7 and +16..+23
    const _Float16* xr0 = xb + (size_t)(m0 + l15) * K_PAD + hi * 8;
    const _Float16* xr1 = xr0 + (size_t)16 * K_PAD;
    // B fragment: col j = n0 + nt*16 + l15; K = kk+hi*8..+7 and +16..+23
    const _Float16* br  = w1b + (size_t)(n0 + l15) * K_PAD + hi * 8;

    for (int kk = 0; kk < K_PAD; kk += 32) {
        AFrag a0, a1;
        a0.h2[0] = *(const v8h*)(xr0 + kk);
        a0.h2[1] = *(const v8h*)(xr0 + kk + 16);
        a1.h2[0] = *(const v8h*)(xr1 + kk);
        a1.h2[1] = *(const v8h*)(xr1 + kk + 16);
        #pragma unroll
        for (int nt = 0; nt < 4; ++nt) {
            AFrag b;
            const _Float16* bp = br + (size_t)nt * 16 * K_PAD + kk;
            b.h2[0] = *(const v8h*)(bp);
            b.h2[1] = *(const v8h*)(bp + 16);
            acc[0][nt] = __builtin_amdgcn_wmma_f32_16x16x32_f16(
                false, a0.v, false, b.v, (short)0, acc[0][nt], false, false);
            acc[1][nt] = __builtin_amdgcn_wmma_f32_16x16x32_f16(
                false, a1.v, false, b.v, (short)0, acc[1][nt], false, false);
        }
    }

    // Store h tile (D layout: lane<16 -> rows 0..7, lane>=16 -> rows 8..15, col = l15)
    #pragma unroll
    for (int mt = 0; mt < 2; ++mt) {
        int rbase = m0 + mt * 16 + hi * 8;
        #pragma unroll
        for (int nt = 0; nt < 4; ++nt) {
            int col = n0 + nt * 16 + l15;
            #pragma unroll
            for (int r = 0; r < 8; ++r)
                h[(size_t)(rbase + r) * D_H + col] = acc[mt][nt][r];
        }
    }

    // Per-column partial sums (32 rows of this block-strip) -> global atomics
    #pragma unroll
    for (int nt = 0; nt < 4; ++nt) {
        float s = 0.f, q = 0.f;
        #pragma unroll
        for (int mt = 0; mt < 2; ++mt)
            #pragma unroll
            for (int r = 0; r < 8; ++r) {
                float v = acc[mt][nt][r];
                s += v; q += v * v;
            }
        s += __shfl_xor(s, 16, 32);   // combine lane pair sharing the same column
        q += __shfl_xor(q, 16, 32);
        if (lane < 16) {
            atomicAdd(&colsum[n0 + nt * 16 + lane], s);
            atomicAdd(&colsq [n0 + nt * 16 + lane], q);
        }
    }
}

// ---------------- BN finalize: scale/shift per hidden column ----------------

__global__ void bn_finalize_kernel(const float* __restrict__ colsum, const float* __restrict__ colsq,
                                   const float* __restrict__ gamma, const float* __restrict__ beta,
                                   float* __restrict__ scale, float* __restrict__ shift)
{
    int j = blockIdx.x * blockDim.x + threadIdx.x;
    if (j < D_H) {
        const float invB = 1.0f / (float)B_ROWS;
        float mu  = colsum[j] * invB;
        float var = colsq[j] * invB - mu * mu;
        float inv = rsqrtf(var + 1e-5f);
        float sc  = gamma[j] * inv;
        scale[j] = sc;
        shift[j] = beta[j] - mu * sc;
    }
}

// ---------------- GEMM2: out = sign(h*scale+shift) @ sign(w2)^T ----------------

union ABits { v16h v; unsigned short u[16]; };

__device__ inline void sign4(const float4 f, const float4 sc, const float4 sh,
                             unsigned short* u) {
    u[0] = (f.x * sc.x + sh.x) < 0.f ? (unsigned short)0xBC00u : (unsigned short)0x3C00u;
    u[1] = (f.y * sc.y + sh.y) < 0.f ? (unsigned short)0xBC00u : (unsigned short)0x3C00u;
    u[2] = (f.z * sc.z + sh.z) < 0.f ? (unsigned short)0xBC00u : (unsigned short)0x3C00u;
    u[3] = (f.w * sc.w + sh.w) < 0.f ? (unsigned short)0xBC00u : (unsigned short)0x3C00u;
}

__global__ __launch_bounds__(256)
void gemm2_wmma_kernel(const float* __restrict__ h, const float* __restrict__ scale,
                       const float* __restrict__ shift, const _Float16* __restrict__ w2b,
                       float* __restrict__ out)
{
    const int lane = threadIdx.x & 31;
    const int wave = threadIdx.x >> 5;
    const int l15  = lane & 15;
    const int hi   = lane >> 4;
    const int m0   = (blockIdx.x * 8 + wave) * 16;   // 16-row tile per wave

    v8f acc = {};
    const float*    hrow = h   + (size_t)(m0 + l15) * D_H;
    const _Float16* brow = w2b + (size_t)l15 * D_H + hi * 8;

    for (int kk = 0; kk < D_H; kk += 32) {
        int base = kk + hi * 8;
        float4 f0 = *(const float4*)(hrow + base);
        float4 f1 = *(const float4*)(hrow + base + 4);
        float4 f2 = *(const float4*)(hrow + base + 16);
        float4 f3 = *(const float4*)(hrow + base + 20);
        float4 s0 = *(const float4*)(scale + base);
        float4 s1 = *(const float4*)(scale + base + 4);
        float4 s2 = *(const float4*)(scale + base + 16);
        float4 s3 = *(const float4*)(scale + base + 20);
        float4 t0 = *(const float4*)(shift + base);
        float4 t1 = *(const float4*)(shift + base + 4);
        float4 t2 = *(const float4*)(shift + base + 16);
        float4 t3 = *(const float4*)(shift + base + 20);

        ABits a;
        sign4(f0, s0, t0, a.u + 0);
        sign4(f1, s1, t1, a.u + 4);
        sign4(f2, s2, t2, a.u + 8);
        sign4(f3, s3, t3, a.u + 12);

        AFrag b;
        b.h2[0] = *(const v8h*)(brow + kk);
        b.h2[1] = *(const v8h*)(brow + kk + 16);

        acc = __builtin_amdgcn_wmma_f32_16x16x32_f16(
            false, a.v, false, b.v, (short)0, acc, false, false);
    }

    // store only real classes (cols 0..9); padded cols 10..15 discarded
    if (l15 < N_CLS) {
        int rbase = m0 + hi * 8;
        #pragma unroll
        for (int r = 0; r < 8; ++r)
            out[(size_t)(rbase + r) * N_CLS + l15] = acc[r];
    }
}

// ---------------- launch ----------------

extern "C" void kernel_launch(void* const* d_in, const int* in_sizes, int n_in,
                              void* d_out, int out_size, void* d_ws, size_t ws_size,
                              hipStream_t stream) {
    const float* x     = (const float*)d_in[0];
    const float* w1    = (const float*)d_in[1];
    const float* gamma = (const float*)d_in[2];
    const float* beta  = (const float*)d_in[3];
    const float* w2    = (const float*)d_in[4];
    float* out = (float*)d_out;

    char* ws = (char*)d_ws;
    size_t off = 0;
    _Float16* xb  = (_Float16*)(ws + off); off += (size_t)B_ROWS * K_PAD * 2;   // 104,857,600
    _Float16* w1b = (_Float16*)(ws + off); off += (size_t)D_H * K_PAD * 2;      //   1,638,400
    _Float16* w2b = (_Float16*)(ws + off); off += (size_t)16 * D_H * 2;         //      32,768
    float* h      = (float*)(ws + off);    off += (size_t)B_ROWS * D_H * 4;     // 268,435,456
    float* colsum = (float*)(ws + off);    off += 4096;
    float* colsq  = (float*)(ws + off);    off += 4096;
    float* scale  = (float*)(ws + off);    off += 4096;
    float* shift  = (float*)(ws + off);    off += 4096;

    prep_x_kernel <<<dim3(B_ROWS, 4), 256, 0, stream>>>(x,  xb);
    prep_w1_kernel<<<dim3(D_H,    4), 256, 0, stream>>>(w1, w1b);
    prep_w2_kernel<<<dim3(16,     4), 256, 0, stream>>>(w2, w2b);
    zero_stats_kernel<<<4, 256, 0, stream>>>(colsum, colsq);

    gemm1_wmma_kernel<<<dim3(B_ROWS / 32, 2), 256, 0, stream>>>(xb, w1b, h, colsum, colsq);

    bn_finalize_kernel<<<4, 256, 0, stream>>>(colsum, colsq, gamma, beta, scale, shift);

    gemm2_wmma_kernel<<<B_ROWS / 16 / 8, 256, 0, stream>>>(h, scale, shift, w2b, out);
}
